// Direction_17549236371980
// MI455X (gfx1250) — matrix operand order repaired and checked by reading, not compiled, not run on hardware
//
#include <hip/hip_runtime.h>
#include <hip/hip_bf16.h>
#include <math.h>

typedef __bf16 bf16_t;
typedef __attribute__((ext_vector_type(16))) __bf16 v16bf;
typedef __attribute__((ext_vector_type(8)))  float  v8f;

union Frag { v16bf v; unsigned u[8]; };

// ---------------- problem constants ----------------
constexpr int kNV   = 6;          // camera views
constexpr int kDIM  = 128;
constexpr int kFD   = 256;
constexpr int kFH   = 24;
constexpr int kFW   = 60;
constexpr int kHW   = kFH * kFW;  // 1440
constexpr int kQ    = 625;        // 25*25
constexpr int kQP   = 640;        // Q padded to /32
constexpr int kTOK  = kNV * kHW;  // 8640
constexpr int kHEADS= 8;
constexpr int kDH   = 64;
constexpr int kHD   = kHEADS * kDH; // 512

// =====================================================================
// WMMA GEMM:  C[M,N] (f32 accum) = A[M,K] * B[K,N]
//   A: row-major, element (m,k) = A[m*ldaM + k]  (contiguous K)
//   B: bTrans==0: element (k,n) = B[k*ldb + n]   (contiguous N)
//      bTrans==1: element (k,n) = B[n*ldb + k]   (contiguous K, i.e. N x K)
// Block = 128 threads (4 waves); block tile 32(M) x 64(N); wave w owns
// columns [w*16, w*16+16) and BOTH 16-row halves (2 accumulators, B reuse).
// K stepped by 32, LDS double-buffered. LDS layouts give packed gathers:
//   As[m][k] (k fastest, 32x32), Bs[n][k] (k fastest, 64x32).
// Epilogue: optional bias[n], optional strided f32 addend, f32/bf16 output.
// Requires: M%32==0, N%64==0, K%32==0, 16B-aligned A rows (callers ensure).
// =====================================================================
__global__ void gemm_wmma_bf16(
    const bf16_t* __restrict__ A, long ldaM,
    const bf16_t* __restrict__ Bm, long ldb, int bTrans,
    int M, int N, int K,
    float* __restrict__ Cf, bf16_t* __restrict__ Cb, long ldc,
    const float* __restrict__ bias,
    const float* __restrict__ add, long addRS, long addCS)
{
    __shared__ bf16_t As[2][32 * 32];   // [m][k]
    __shared__ bf16_t Bs[2][64 * 32];   // [n][k]

    const int tid  = threadIdx.x;      // 0..127
    const int wave = tid >> 5;         // 0..3
    const int lane = tid & 31;
    const int row0 = blockIdx.y * 32;
    const int col0 = blockIdx.x * 64;

    const int h  = lane >> 4;          // half-wave select
    const int mr = lane & 15;          // row within 16-tile / col within 16-tile
    const int nc = wave * 16 + mr;     // column within 64-wide block tile

    v8f acc0 = {0.f, 0.f, 0.f, 0.f, 0.f, 0.f, 0.f, 0.f};
    v8f acc1 = {0.f, 0.f, 0.f, 0.f, 0.f, 0.f, 0.f, 0.f};

    auto stageA = [&](int buf, int kk) {
        // 32x32 tile = 1024 elems; one 16B chunk per thread
        int m  = tid >> 2;             // 0..31
        int k8 = (tid & 3) * 8;        // 0,8,16,24
        *(float4*)&As[buf][m * 32 + k8] =
            *(const float4*)(A + (long)(row0 + m) * ldaM + (kk + k8));
    };
    auto stageB = [&](int buf, int kk) {
        if (bTrans) {
            // global contiguous along K -> 16B loads, b128 LDS stores
            #pragma unroll
            for (int i = 0; i < 2; ++i) {
                int ch = tid + i * 128;        // 0..255
                int n  = ch >> 2;              // 0..63
                int k8 = (ch & 3) * 8;
                *(float4*)&Bs[buf][n * 32 + k8] =
                    *(const float4*)(Bm + (long)(col0 + n) * ldb + (kk + k8));
            }
        } else {
            // global contiguous along N -> wide load, 4-way split store
            #pragma unroll
            for (int i = 0; i < 4; ++i) {
                int ch = tid + i * 128;        // 0..511
                int k  = ch >> 4;              // 0..31
                int n4 = (ch & 15) * 4;
                unsigned long long w =
                    *(const unsigned long long*)(Bm + (long)(kk + k) * ldb + (col0 + n4));
                const bf16_t* e = (const bf16_t*)&w;
                Bs[buf][(n4 + 0) * 32 + k] = e[0];
                Bs[buf][(n4 + 1) * 32 + k] = e[1];
                Bs[buf][(n4 + 2) * 32 + k] = e[2];
                Bs[buf][(n4 + 3) * 32 + k] = e[3];
            }
        }
    };

    stageA(0, 0);
    stageB(0, 0);
    int p = 0;
    for (int kk = 0; kk < K; kk += 32) {
        __syncthreads();
        if (kk + 32 < K) { stageA(p ^ 1, kk + 32); stageB(p ^ 1, kk + 32); }

        const unsigned* As32 = (const unsigned*)As[p];
        const unsigned* Bs32 = (const unsigned*)Bs[p];
        // A fragment (16-bit A 16x32): word v covers K = 16h+2v, 16h+2v+1
        // word index h*8 + v within a 16-word LDS row
        Frag af0, af1, bfr;
        #pragma unroll
        for (int v = 0; v < 8; ++v) {
            int wi = h * 8 + v;
            af0.u[v] = As32[mr * 16 + wi];            // rows 0..15 of tile
            af1.u[v] = As32[(16 + mr) * 16 + wi];     // rows 16..31 of tile
        }
        // B fragment (32x16): lanes 0-15 K=0..15, lanes 16-31 K=16..31
        #pragma unroll
        for (int w = 0; w < 8; ++w)
            bfr.u[w] = Bs32[nc * 16 + h * 8 + w];

        acc0 = __builtin_amdgcn_wmma_f32_16x16x32_bf16(
            false, af0.v, false, bfr.v, (short)0, acc0, false, false);
        acc1 = __builtin_amdgcn_wmma_f32_16x16x32_bf16(
            false, af1.v, false, bfr.v, (short)0, acc1, false, false);
        p ^= 1;
    }

    // ---- store: acc0 row r -> C[row0 + r + 8h][nc]; acc1 -> +16 rows ----
    const int cn = col0 + nc;
    #pragma unroll
    for (int r = 0; r < 8; ++r) {
        {
            int cm = row0 + r + 8 * h;
            float val = acc0[r];
            if (bias) val += bias[cn];
            if (add)  val += add[(long)cm * addRS + (long)cn * addCS];
            if (Cf) Cf[(long)cm * ldc + cn] = val;
            if (Cb) Cb[(long)cm * ldc + cn] = (bf16_t)val;
        }
        {
            int cm = row0 + 16 + r + 8 * h;
            float val = acc1[r];
            if (bias) val += bias[cn];
            if (add)  val += add[(long)cm * addRS + (long)cn * addCS];
            if (Cf) Cf[(long)cm * ldc + cn] = val;
            if (Cb) Cb[(long)cm * ldc + cn] = (bf16_t)val;
        }
    }
}

// =====================================================================
// Geometry: c_embed[n][d] and normalized img_embed[n][d][k]
// =====================================================================
__global__ void geom_kernel(const float* __restrict__ I_inv,
                            const float* __restrict__ E_inv,
                            const float* __restrict__ cam_w,
                            const float* __restrict__ img_w,
                            float* __restrict__ c_embed,
                            float* __restrict__ img_embed)
{
    int t = blockIdx.x * blockDim.x + threadIdx.x;
    if (t >= kNV * kHW) return;
    int n = t / kHW, k = t % kHW;
    int i = k / kFW, j = k % kFW;
    float px = ((float)j / (float)(kFW - 1)) * 480.f;
    float py = ((float)i / (float)(kFH - 1)) * 224.f;

    const float* Iv = I_inv + n * 9;
    float c0 = Iv[0] * px + Iv[1] * py + Iv[2];
    float c1 = Iv[3] * px + Iv[4] * py + Iv[5];
    float c2 = Iv[6] * px + Iv[7] * py + Iv[8];

    const float* Ev = E_inv + n * 16;
    float dv[4], e3[4];
    #pragma unroll
    for (int c = 0; c < 4; ++c) {
        dv[c] = Ev[c * 4 + 0] * c0 + Ev[c * 4 + 1] * c1 + Ev[c * 4 + 2] * c2 + Ev[c * 4 + 3];
        e3[c] = Ev[c * 4 + 3];
    }

    float nrm = 0.f;
    for (int d = 0; d < kDIM; ++d) {
        const float* iw = img_w + d * 4;
        const float* cw = cam_w + d * 4;
        float ce = e3[0] * cw[0] + e3[1] * cw[1] + e3[2] * cw[2] + e3[3] * cw[3];
        float e  = iw[0] * dv[0] + iw[1] * dv[1] + iw[2] * dv[2] + iw[3] * dv[3] - ce;
        nrm += e * e;
        if (k == 0) c_embed[n * kDIM + d] = ce;
    }
    float inv = 1.f / (sqrtf(nrm) + 1e-7f);
    for (int d = 0; d < kDIM; ++d) {
        const float* iw = img_w + d * 4;
        const float* cw = cam_w + d * 4;
        float ce = e3[0] * cw[0] + e3[1] * cw[1] + e3[2] * cw[2] + e3[3] * cw[3];
        float e  = iw[0] * dv[0] + iw[1] * dv[1] + iw[2] * dv[2] + iw[3] * dv[3] - ce;
        img_embed[((long)n * kDIM + d) * kHW + k] = e * inv;
    }
}

// =====================================================================
// BEV embed + query (views 0,1 only): query[n][q][d] = norm(bev_e) + x[d][q]
// =====================================================================
__global__ void bev_query_kernel(const float* __restrict__ bev_w,
                                 const float* __restrict__ bev_b,
                                 const float* __restrict__ c_embed,
                                 const float* __restrict__ x,
                                 float* __restrict__ query01)
{
    int t = blockIdx.x * blockDim.x + threadIdx.x;
    if (t >= 2 * kQ) return;
    int n = t / kQ, q = t % kQ;
    int i = q / 25, j = q % 25;
    float u = ((float)j / 24.f) * 200.f;   // gb0 * BEV_W
    float v = ((float)i / 24.f) * 200.f;   // gb1 * BEV_H
    // V^{-1}: wx = (BEV_H*OFF + BEV_H/2 - v)/sh ; wy = (BEV_W/2 - u)/sw ; sh=sw=2
    float wx = (100.f - v) * 0.5f;
    float wy = (100.f - u) * 0.5f;

    const float* ce = c_embed + n * kDIM;
    float nrm = 0.f;
    for (int d = 0; d < kDIM; ++d) {
        float e = bev_w[d * 2] * wx + bev_w[d * 2 + 1] * wy + bev_b[d] - ce[d];
        nrm += e * e;
    }
    float inv = 1.f / (sqrtf(nrm) + 1e-7f);
    float* out = query01 + ((long)n * kQP + q) * kDIM;
    for (int d = 0; d < kDIM; ++d) {
        float e = bev_w[d * 2] * wx + bev_w[d * 2 + 1] * wy + bev_b[d] - ce[d];
        out[d] = e * inv + x[d * kQ + q];
    }
}

// =====================================================================
// BN(eval) + ReLU -> bf16, transposed to token-major A[br][n][k][c]
// =====================================================================
__global__ void bnrelu_kernel(const float* __restrict__ feat,
                              const float* g0, const float* b0, const float* m0, const float* v0,
                              const float* g1, const float* b1, const float* m1, const float* v1,
                              bf16_t* __restrict__ Abn)
{
    long t = (long)blockIdx.x * blockDim.x + threadIdx.x;
    const long total = 2L * kNV * kHW * kFD;
    if (t >= total) return;
    int c  = (int)(t & (kFD - 1));
    long r = t >> 8;
    int k  = (int)(r % kHW);
    long r2 = r / kHW;
    int n  = (int)(r2 % kNV);
    int br = (int)(r2 / kNV);
    float f = feat[((long)n * kFD + c) * kHW + k];
    float gg = br ? g1[c] : g0[c];
    float bb = br ? b1[c] : b0[c];
    float mm = br ? m1[c] : m0[c];
    float vv = br ? v1[c] : v0[c];
    float val = (f - mm) * rsqrtf(vv + 1e-5f) * gg + bb;
    val = fmaxf(val, 0.f);
    Abn[t] = (bf16_t)val;
}

// transpose f32 [R][C] -> bf16 [C][R]
__global__ void transpose_bf16_kernel(const float* __restrict__ src,
                                      bf16_t* __restrict__ dst, int R, int C)
{
    int t = blockIdx.x * blockDim.x + threadIdx.x;
    if (t >= R * C) return;
    int r = t / C, c = t % C;
    dst[(long)c * R + r] = (bf16_t)src[(long)r * C + c];
}

// =====================================================================
// LayerNorm over rows of [rows][128] f32 -> bf16 (one wave per row)
// =====================================================================
__global__ void ln_rows_kernel(const float* __restrict__ src,
                               const float* __restrict__ g,
                               const float* __restrict__ b,
                               bf16_t* __restrict__ dst, int rows)
{
    int r = blockIdx.x;
    if (r >= rows) return;
    int lane = threadIdx.x;   // 0..31
    const float* p = src + (long)r * kDIM;
    float xv[4];
    float s = 0.f;
    #pragma unroll
    for (int t = 0; t < 4; ++t) { xv[t] = p[lane + 32 * t]; s += xv[t]; }
    #pragma unroll
    for (int off = 16; off >= 1; off >>= 1) s += __shfl_xor(s, off, 32);
    float mu = s / (float)kDIM;
    float vs = 0.f;
    #pragma unroll
    for (int t = 0; t < 4; ++t) { float d = xv[t] - mu; vs += d * d; }
    #pragma unroll
    for (int off = 16; off >= 1; off >>= 1) vs += __shfl_xor(vs, off, 32);
    float inv = rsqrtf(vs / (float)kDIM + 1e-5f);
    #pragma unroll
    for (int t = 0; t < 4; ++t) {
        int d = lane + 32 * t;
        dst[(long)r * kDIM + d] = (bf16_t)((xv[t] - mu) * inv * g[d] + b[d]);
    }
}

// Qs[q][o] = sc * (Qp[view0 row q][o] + Qp[view1 row q][o])   (bf16)
__global__ void qsum_kernel(const bf16_t* __restrict__ Qp, bf16_t* __restrict__ Qs)
{
    int t = blockIdx.x * blockDim.x + threadIdx.x;
    if (t >= kQP * kHD) return;
    int q = t / kHD, o = t % kHD;
    float a = (float)Qp[(long)q * kHD + o];
    float c = (float)Qp[(long)(kQP + q) * kHD + o];
    Qs[t] = (bf16_t)(0.125f * (a + c));   // sc = DH^-0.5 = 0.125
}

// row softmax of S[row][TOK] -> bf16 P
__global__ void softmax_rows_kernel(float* __restrict__ S, bf16_t* __restrict__ P)
{
    int r = blockIdx.x;
    float* row = S + (long)r * kTOK;
    __shared__ float red[256];
    int tid = threadIdx.x;
    float mx = -1e30f;
    for (int j = tid; j < kTOK; j += 256) mx = fmaxf(mx, row[j]);
    red[tid] = mx; __syncthreads();
    for (int s = 128; s > 0; s >>= 1) { if (tid < s) red[tid] = fmaxf(red[tid], red[tid + s]); __syncthreads(); }
    mx = red[0]; __syncthreads();
    float sum = 0.f;
    for (int j = tid; j < kTOK; j += 256) { float e = expf(row[j] - mx); row[j] = e; sum += e; }
    red[tid] = sum; __syncthreads();
    for (int s = 128; s > 0; s >>= 1) { if (tid < s) red[tid] += red[tid + s]; __syncthreads(); }
    float inv = 1.f / red[0];
    bf16_t* pr = P + (long)r * kTOK;
    for (int j = tid; j < kTOK; j += 256) pr[j] = (bf16_t)(row[j] * inv);
}

// z1 = LN( Obuf @ proj_w^T + proj_b + x^T , pre ) ; one block(128) per q row
__global__ void out_proj_ln_kernel(const float* __restrict__ Obuf,
                                   const float* __restrict__ proj_w,
                                   const float* __restrict__ proj_b,
                                   const float* __restrict__ x,
                                   const float* __restrict__ pre_g,
                                   const float* __restrict__ pre_b,
                                   float* __restrict__ z1)
{
    int q = blockIdx.x;
    int d = threadIdx.x;   // 0..127
    __shared__ float orow[kHD];
    for (int t = d; t < kHD; t += kDIM) orow[t] = Obuf[(long)q * kHD + t];
    __syncthreads();
    float val = proj_b[d] + x[d * kQ + q];
    const float* w = proj_w + (long)d * kHD;
    for (int o = 0; o < kHD; ++o) val += orow[o] * w[o];
    __shared__ float red[kDIM];
    red[d] = val; __syncthreads();
    for (int s = 64; s > 0; s >>= 1) { if (d < s) red[d] += red[d + s]; __syncthreads(); }
    float mu = red[0] / (float)kDIM; __syncthreads();
    float df = val - mu;
    red[d] = df * df; __syncthreads();
    for (int s = 64; s > 0; s >>= 1) { if (d < s) red[d] += red[d + s]; __syncthreads(); }
    float var = red[0] / (float)kDIM;
    z1[(long)q * kDIM + d] = df * rsqrtf(var + 1e-5f) * pre_g[d] + pre_b[d];
}

// h1 = gelu_exact(z1 @ mlp1_w^T + b) ; block(256) per q row
__global__ void mlp1_kernel(const float* __restrict__ z1,
                            const float* __restrict__ w,
                            const float* __restrict__ b,
                            float* __restrict__ h1)
{
    int q = blockIdx.x;
    int e = threadIdx.x;   // 0..255
    __shared__ float zr[kDIM];
    if (e < kDIM) zr[e] = z1[(long)q * kDIM + e];
    __syncthreads();
    float s = b[e];
    const float* wr = w + (long)e * kDIM;
    for (int d = 0; d < kDIM; ++d) s += zr[d] * wr[d];
    h1[(long)q * (2 * kDIM) + e] = 0.5f * s * (1.f + erff(s * 0.70710678118654752f));
}

// out = LN( z1 + h1 @ mlp2_w^T + b , post ), transposed store ; block(128)/row
__global__ void mlp2_ln_out_kernel(const float* __restrict__ z1,
                                   const float* __restrict__ h1,
                                   const float* __restrict__ w,
                                   const float* __restrict__ b,
                                   const float* __restrict__ post_g,
                                   const float* __restrict__ post_b,
                                   float* __restrict__ out)
{
    int q = blockIdx.x;
    int d = threadIdx.x;   // 0..127
    __shared__ float hr[2 * kDIM];
    for (int t = d; t < 2 * kDIM; t += kDIM) hr[t] = h1[(long)q * (2 * kDIM) + t];
    __syncthreads();
    float s = b[d];
    const float* wr = w + (long)d * (2 * kDIM);
    for (int e = 0; e < 2 * kDIM; ++e) s += hr[e] * wr[e];
    float val = z1[(long)q * kDIM + d] + s;
    __shared__ float red[kDIM];
    red[d] = val; __syncthreads();
    for (int ss = 64; ss > 0; ss >>= 1) { if (d < ss) red[d] += red[d + ss]; __syncthreads(); }
    float mu = red[0] / (float)kDIM; __syncthreads();
    float df = val - mu;
    red[d] = df * df; __syncthreads();
    for (int ss = 64; ss > 0; ss >>= 1) { if (d < ss) red[d] += red[d + ss]; __syncthreads(); }
    float var = red[0] / (float)kDIM;
    out[d * kQ + q] = df * rsqrtf(var + 1e-5f) * post_g[d] + post_b[d];
}

// =====================================================================
extern "C" void kernel_launch(void* const* d_in, const int* in_sizes, int n_in,
                              void* d_out, int out_size, void* d_ws, size_t ws_size,
                              hipStream_t stream)
{
    // ---- inputs (setup_inputs dict order, params flattened) ----
    const float* x       = (const float*)d_in[0];
    const float* feat    = (const float*)d_in[1];
    const float* I_inv   = (const float*)d_in[2];
    const float* E_inv   = (const float*)d_in[3];
    const float* cam_w   = (const float*)d_in[4];
    const float* img_w   = (const float*)d_in[5];
    const float* bev_w   = (const float*)d_in[6];
    const float* bev_b   = (const float*)d_in[7];
    const float* pj_bn_g = (const float*)d_in[8];
    const float* pj_bn_b = (const float*)d_in[9];
    const float* pj_bn_m = (const float*)d_in[10];
    const float* pj_bn_v = (const float*)d_in[11];
    const float* pj_cw   = (const float*)d_in[12];
    const float* ln_bn_g = (const float*)d_in[13];
    const float* ln_bn_b = (const float*)d_in[14];
    const float* ln_bn_m = (const float*)d_in[15];
    const float* ln_bn_v = (const float*)d_in[16];
    const float* ln_cw   = (const float*)d_in[17];
    const float* q_ln_g  = (const float*)d_in[18];
    const float* q_ln_b  = (const float*)d_in[19];
    const float* q_w     = (const float*)d_in[20];
    const float* q_b     = (const float*)d_in[21];
    const float* k_ln_g  = (const float*)d_in[22];
    const float* k_ln_b  = (const float*)d_in[23];
    const float* k_w     = (const float*)d_in[24];
    const float* k_b     = (const float*)d_in[25];
    const float* v_ln_g  = (const float*)d_in[26];
    const float* v_ln_b  = (const float*)d_in[27];
    const float* v_w     = (const float*)d_in[28];
    const float* v_b     = (const float*)d_in[29];
    const float* proj_w  = (const float*)d_in[30];
    const float* proj_b  = (const float*)d_in[31];
    const float* pre_g   = (const float*)d_in[32];
    const float* pre_b   = (const float*)d_in[33];
    const float* mlp1_w  = (const float*)d_in[34];
    const float* mlp1_b  = (const float*)d_in[35];
    const float* mlp2_w  = (const float*)d_in[36];
    const float* mlp2_b  = (const float*)d_in[37];
    const float* post_g  = (const float*)d_in[38];
    const float* post_b  = (const float*)d_in[39];
    float* out = (float*)d_out;

    // ---- workspace carve-up (all chunks 256B-aligned) ----
    char* ws = (char*)d_ws;
    size_t off = 0;
    auto carve = [&](size_t bytes) { size_t o = off; off += (bytes + 255) & ~(size_t)255; return o; };
    float*  c_embed  = (float*)(ws + carve(kNV * kDIM * 4));
    float*  img_emb  = (float*)(ws + carve((size_t)kNV * kDIM * kHW * 4));
    float*  query01  = (float*)(ws + carve((size_t)2 * kQP * kDIM * 4));
    float*  key_tok  = (float*)(ws + carve((size_t)kTOK * kDIM * 4));
    float*  val_tok  = (float*)(ws + carve((size_t)kTOK * kDIM * 4));
    float*  Sbuf     = (float*)(ws + carve((size_t)kQP * kTOK * 4));
    float*  Obuf     = (float*)(ws + carve((size_t)kQP * kHD * 4));
    float*  z1       = (float*)(ws + carve((size_t)kQP * kDIM * 4));
    float*  h1       = (float*)(ws + carve((size_t)kQP * 2 * kDIM * 4));
    bf16_t* Abn      = (bf16_t*)(ws + carve((size_t)2 * kNV * kHW * kFD * 2));
    bf16_t* Wc       = (bf16_t*)(ws + carve((size_t)2 * kFD * kDIM * 2));   // [br][c][d]
    bf16_t* qwb      = (bf16_t*)(ws + carve((size_t)kDIM * kHD * 2));
    bf16_t* kwb      = (bf16_t*)(ws + carve((size_t)kDIM * kHD * 2));
    bf16_t* vwb      = (bf16_t*)(ws + carve((size_t)kDIM * kHD * 2));
    bf16_t* lnQ      = (bf16_t*)(ws + carve((size_t)2 * kQP * kDIM * 2));
    bf16_t* lnK      = (bf16_t*)(ws + carve((size_t)kTOK * kDIM * 2));
    bf16_t* lnV      = (bf16_t*)(ws + carve((size_t)kTOK * kDIM * 2));
    bf16_t* Qp       = (bf16_t*)(ws + carve((size_t)2 * kQP * kHD * 2));
    bf16_t* Qs       = (bf16_t*)(ws + carve((size_t)kQP * kHD * 2));
    bf16_t* Kmat     = (bf16_t*)(ws + carve((size_t)kTOK * kHD * 2));
    bf16_t* Vmat     = (bf16_t*)(ws + carve((size_t)kTOK * kHD * 2));
    bf16_t* Pb       = (bf16_t*)(ws + carve((size_t)kQP * kTOK * 2));
    size_t total = off;

    // zero workspace (padding rows must be 0; harness poisons with 0xAA)
    hipMemsetAsync(d_ws, 0, total, stream);

    auto gemm = [&](const bf16_t* A, long laM,
                    const bf16_t* B, long lb, int bT,
                    int M, int N, int K,
                    float* Cf, bf16_t* Cb, long ldc,
                    const float* bias, const float* add, long aRS, long aCS) {
        dim3 g(N / 64, M / 32);
        gemm_wmma_bf16<<<g, dim3(128), 0, stream>>>(A, laM, B, lb, bT,
                                                    M, N, K, Cf, Cb, ldc, bias, add, aRS, aCS);
    };

    // 1) geometry + BEV query
    geom_kernel<<<(kNV * kHW + 127) / 128, 128, 0, stream>>>(I_inv, E_inv, cam_w, img_w,
                                                             c_embed, img_emb);
    bev_query_kernel<<<(2 * kQ + 127) / 128, 128, 0, stream>>>(bev_w, bev_b, c_embed, x, query01);

    // 2) BN + ReLU -> bf16 token-major
    {
        long tot = 2L * kNV * kHW * kFD;
        bnrelu_kernel<<<(unsigned)((tot + 255) / 256), 256, 0, stream>>>(
            feat, pj_bn_g, pj_bn_b, pj_bn_m, pj_bn_v,
            ln_bn_g, ln_bn_b, ln_bn_m, ln_bn_v, Abn);
    }

    // 3) weight transposes -> bf16 B matrices
    transpose_bf16_kernel<<<(kDIM * kFD + 255) / 256, 256, 0, stream>>>(pj_cw, Wc,              kDIM, kFD);
    transpose_bf16_kernel<<<(kDIM * kFD + 255) / 256, 256, 0, stream>>>(ln_cw, Wc + kFD * kDIM, kDIM, kFD);
    transpose_bf16_kernel<<<(kHD * kDIM + 255) / 256, 256, 0, stream>>>(q_w, qwb, kHD, kDIM);
    transpose_bf16_kernel<<<(kHD * kDIM + 255) / 256, 256, 0, stream>>>(k_w, kwb, kHD, kDIM);
    transpose_bf16_kernel<<<(kHD * kDIM + 255) / 256, 256, 0, stream>>>(v_w, vwb, kHD, kDIM);

    // 4) conv GEMMs: key_tok[n][k][d] = Abn0 @ Wc0 + img_emb^T ; val_tok = Abn1 @ Wc1
    for (int n = 0; n < kNV; ++n) {
        const bf16_t* A0 = Abn + (long)n * kHW * kFD;
        const bf16_t* A1 = Abn + ((long)kNV + n) * kHW * kFD;
        gemm(A0, kFD, Wc, kDIM, 0, kHW, kDIM, kFD,
             key_tok + (long)n * kHW * kDIM, nullptr, kDIM,
             nullptr, img_emb + (long)n * kDIM * kHW, 1, kHW);
        gemm(A1, kFD, Wc + kFD * kDIM, kDIM, 0, kHW, kDIM, kFD,
             val_tok + (long)n * kHW * kDIM, nullptr, kDIM,
             nullptr, nullptr, 0, 0);
    }

    // 5) LayerNorms -> bf16
    ln_rows_kernel<<<2 * kQP, 32, 0, stream>>>(query01, q_ln_g, q_ln_b, lnQ, 2 * kQP);
    ln_rows_kernel<<<kTOK, 32, 0, stream>>>(key_tok, k_ln_g, k_ln_b, lnK, kTOK);
    ln_rows_kernel<<<kTOK, 32, 0, stream>>>(val_tok, v_ln_g, v_ln_b, lnV, kTOK);

    // 6) q/k/v projection GEMMs (bf16 out, +bias)
    gemm(lnQ, kDIM, qwb, kHD, 0, 2 * kQP, kHD, kDIM, nullptr, Qp,   kHD, q_b, nullptr, 0, 0);
    gemm(lnK, kDIM, kwb, kHD, 0, kTOK,    kHD, kDIM, nullptr, Kmat, kHD, k_b, nullptr, 0, 0);
    gemm(lnV, kDIM, vwb, kHD, 0, kTOK,    kHD, kDIM, nullptr, Vmat, kHD, v_b, nullptr, 0, 0);

    // 7) fold views 0+1 and scale
    qsum_kernel<<<(kQP * kHD + 255) / 256, 256, 0, stream>>>(Qp, Qs);

    // 8) attention per head: S = Qs_m @ Kmat_m^T ; softmax ; O_m = P @ Vmat_m
    for (int m = 0; m < kHEADS; ++m) {
        gemm(Qs + m * kDH, kHD,              // A: [QP][64], row stride HD
             Kmat + m * kDH, kHD, 1,         // B: Kmat^T view (contiguous K)
             kQP, kTOK, kDH,
             Sbuf, nullptr, kTOK, nullptr, nullptr, 0, 0);
        softmax_rows_kernel<<<kQP, 256, 0, stream>>>(Sbuf, Pb);
        gemm(Pb, kTOK,                       // A: [QP][TOK]
             Vmat + m * kDH, kHD, 0,         // B: [TOK][64] slice (contiguous N)
             kQP, kDH, kTOK,
             Obuf + m * kDH, nullptr, kHD, nullptr, nullptr, 0, 0);
    }

    // 9) output projection + skip + LN ; MLP ; final LN + transpose store
    out_proj_ln_kernel<<<kQ, kDIM, 0, stream>>>(Obuf, proj_w, proj_b, x, pre_g, pre_b, z1);
    mlp1_kernel<<<kQ, 2 * kDIM, 0, stream>>>(z1, mlp1_w, mlp1_b, h1);
    mlp2_ln_out_kernel<<<kQ, kDIM, 0, stream>>>(z1, h1, mlp2_w, mlp2_b, post_g, post_b, out);

    (void)in_sizes; (void)n_in; (void)out_size; (void)ws_size;
}
